// LinearBase_13451837571091
// MI455X (gfx1250) — 8-bit matrix operand order repaired and checked by reading, compiled, not run on hardware
//
#include <hip/hip_runtime.h>
#include <stdint.h>

typedef int v8i __attribute__((ext_vector_type(8)));
typedef float v4f __attribute__((ext_vector_type(4)));

#define QMAX 127.0f

__global__ __launch_bounds__(256) void quant_rows_kernel(
    const float* __restrict__ x, int8_t* __restrict__ xq,
    float* __restrict__ xscale, int K) {
  const int row = blockIdx.x;
  const int t = threadIdx.x;
  const int nchunk = K >> 10;

  const v4f* xr = reinterpret_cast<const v4f*>(x + (size_t)row * K);
  v4f v[4];
  float amax = 0.0f;
#pragma unroll
  for (int c = 0; c < 4; ++c) {
    if (c < nchunk) {
      v[c] = __builtin_nontemporal_load(&xr[c * 256 + t]);
      amax = fmaxf(amax, fmaxf(fmaxf(fabsf(v[c][0]), fabsf(v[c][1])),
                               fmaxf(fabsf(v[c][2]), fabsf(v[c][3]))));
    }
  }

  __shared__ float red[256];
  red[t] = amax;
  __syncthreads();
#pragma unroll
  for (int s = 128; s > 0; s >>= 1) {
    if (t < s) red[t] = fmaxf(red[t], red[t + s]);
    __syncthreads();
  }
  const float scale = red[0] / QMAX;
  const float inv = 1.0f / fmaxf(scale, 1e-8f);
  if (t == 0) xscale[row] = scale;

  uint32_t* out32 = reinterpret_cast<uint32_t*>(xq + (size_t)row * K);
#pragma unroll
  for (int c = 0; c < 4; ++c) {
    if (c < nchunk) {
      int q0 = (int)fminf(fmaxf(rintf(v[c][0] * inv), -QMAX), QMAX);
      int q1 = (int)fminf(fmaxf(rintf(v[c][1] * inv), -QMAX), QMAX);
      int q2 = (int)fminf(fmaxf(rintf(v[c][2] * inv), -QMAX), QMAX);
      int q3 = (int)fminf(fmaxf(rintf(v[c][3] * inv), -QMAX), QMAX);
      uint32_t pack = (uint32_t)(q0 & 0xff) | ((uint32_t)(q1 & 0xff) << 8) |
                      ((uint32_t)(q2 & 0xff) << 16) |
                      ((uint32_t)(q3 & 0xff) << 24);
      out32[c * 256 + t] = pack;
    }
  }
}

#define ASYNC_B128_GVS(LDSADDR, VOFF, SBASE, OFF)                           \
  asm volatile("global_load_async_to_lds_b128 %0, %1, %2 offset:" #OFF      \
               ::"v"(LDSADDR), "v"(VOFF), "s"(SBASE)                        \
               : "memory")

__device__ __forceinline__ void wait_asynccnt_le12() {
  asm volatile("s_wait_asynccnt 0xc" ::: "memory");
}
__device__ __forceinline__ void wait_asynccnt_0() {
  asm volatile("s_wait_asynccnt 0x0" ::: "memory");
}
__device__ __forceinline__ void sched_fence() {
  asm volatile("" ::: "memory");
}

constexpr int A_STRIDE = 80;
constexpr int B_STRIDE = 80;
constexpr int A_BYTES = 64 * A_STRIDE;
constexpr int B_BYTES = 128 * B_STRIDE;
constexpr int STAGE_BYTES = A_BYTES + B_BYTES;
constexpr int N_BUF = 4;

__global__ __launch_bounds__(128) void gemm_a8w8_wmma_kernel(
    const int8_t* __restrict__ xq, const int8_t* __restrict__ w,
    const float* __restrict__ xscale, const float* __restrict__ wscale,
    float* __restrict__ y, int M, int N, int K) {
  __shared__ __align__(16) uint8_t smem[N_BUF * STAGE_BYTES];

  const int tid = threadIdx.x;
  const int lane = tid & 31;
  const int wave = tid >> 5;
  const int half = lane >> 4;
  const int l = lane & 15;

  const int m0 = blockIdx.y * 64;
  const int n0 = blockIdx.x * 128;
  const int S = K >> 6;

  uint32_t agoff = (uint32_t)((m0 + (tid >> 1)) * K) + (uint32_t)(tid & 1) * 32;
  uint32_t bgoff = (uint32_t)((n0 + tid) * K);
  const uint32_t smem_base = (uint32_t)(uintptr_t)(&smem[0]);
  const uint32_t alds = smem_base + (tid >> 1) * A_STRIDE + (tid & 1) * 32;
  const uint32_t blds = smem_base + A_BYTES + tid * B_STRIDE;

  auto issue_stage = [&](int s) {
    const uint32_t bo = (uint32_t)(s & (N_BUF - 1)) * STAGE_BYTES;
    const uint32_t al = alds + bo;
    const uint32_t bl = blds + bo;
    ASYNC_B128_GVS(al, agoff, xq, 0);
    ASYNC_B128_GVS(al, agoff, xq, 16);
    ASYNC_B128_GVS(bl, bgoff, w, 0);
    ASYNC_B128_GVS(bl, bgoff, w, 16);
    ASYNC_B128_GVS(bl, bgoff, w, 32);
    ASYNC_B128_GVS(bl, bgoff, w, 48);
    agoff += 64;
    bgoff += 64;
  };

  v8i acc[8];
  const v8i vzero = {0, 0, 0, 0, 0, 0, 0, 0};
#pragma unroll
  for (int t = 0; t < 8; ++t) acc[t] = vzero;

  auto compute_stage = [&](int s) {
    const uint8_t* stage = &smem[(s & (N_BUF - 1)) * STAGE_BYTES];

    const uint8_t* sa = stage + (wave * 16 + l) * A_STRIDE + half * 16;
    v8i a;
#pragma unroll
    for (int j = 0; j < 4; ++j) {
      uint2 p = *reinterpret_cast<const uint2*>(sa + (j >> 1) * 32 + (j & 1) * 8);
      a[2 * j] = (int)p.x;
      a[2 * j + 1] = (int)p.y;
    }

    const uint8_t* sb0 = stage + A_BYTES + l * B_STRIDE + half * 16;
    v8i bfr[8];
#pragma unroll
    for (int t = 0; t < 8; ++t) {
      const uint8_t* sb = sb0 + t * 16 * B_STRIDE;
      uint4 q0 = *reinterpret_cast<const uint4*>(sb);
      uint4 q1 = *reinterpret_cast<const uint4*>(sb + 32);
      bfr[t][0] = (int)q0.x; bfr[t][1] = (int)q0.y;
      bfr[t][2] = (int)q0.z; bfr[t][3] = (int)q0.w;
      bfr[t][4] = (int)q1.x; bfr[t][5] = (int)q1.y;
      bfr[t][6] = (int)q1.z; bfr[t][7] = (int)q1.w;
    }
    sched_fence();
#pragma unroll
    for (int t = 0; t < 8; ++t) {
      acc[t] = __builtin_amdgcn_wmma_i32_16x16x64_iu8(true, a, true, bfr[t],
                                                      acc[t], false, false);
    }
  };

  issue_stage(0);
  if (S > 1) issue_stage(1);
  if (S > 2) issue_stage(2);

  int s = 0;
  for (; s + 3 < S; ++s) {
    wait_asynccnt_le12();
    __syncthreads();
    issue_stage(s + 3);
    compute_stage(s);
  }
  for (; s < S; ++s) {
    wait_asynccnt_0();
    __syncthreads();
    compute_stage(s);
  }

  const int mrow = m0 + wave * 16 + half * 8;
  float xs[8];
#pragma unroll
  for (int r = 0; r < 8; ++r) xs[r] = xscale[mrow + r];

#pragma unroll
  for (int t = 0; t < 8; ++t) {
    const int n = n0 + t * 16 + l;
    const float wsc = wscale[n];
#pragma unroll
    for (int r = 0; r < 8; ++r) {
      __builtin_nontemporal_store((float)acc[t][r] * xs[r] * wsc,
                                  &y[(size_t)(mrow + r) * N + n]);
    }
  }
}

extern "C" void kernel_launch(void* const* d_in, const int* in_sizes, int n_in,
                              void* d_out, int out_size, void* d_ws,
                              size_t ws_size, hipStream_t stream) {
  const float* x = (const float*)d_in[0];
  const int8_t* weight = (const int8_t*)d_in[1];
  const float* wscale = (const float*)d_in[2];

  const int N = in_sizes[2];
  const int K = in_sizes[1] / N;
  const int M = in_sizes[0] / K;

  int8_t* xq = (int8_t*)d_ws;
  float* xscale = (float*)((uint8_t*)d_ws + (size_t)M * (size_t)K);

  quant_rows_kernel<<<M, 256, 0, stream>>>(x, xq, xscale, K);

  dim3 grid(N / 128, M / 64);
  gemm_a8w8_wmma_kernel<<<grid, 128, 0, stream>>>(xq, weight, xscale, wscale,
                                                  (float*)d_out, M, N, K);
}
